// Contrast_32487132627462
// MI455X (gfx1250) — matrix operand order repaired and checked by reading, not compiled, not run on hardware
//
#include <hip/hip_runtime.h>

typedef __attribute__((ext_vector_type(16))) _Float16 v16h;
typedef __attribute__((ext_vector_type(8)))  _Float16 v8h;
typedef __attribute__((ext_vector_type(8)))  float    v8f;

#define B_PAIRS  4096
#define DIM      256
#define N2       8192          // 2*B
#define INV_TEMP 2.0f          // 1/0.5

// ---------------------------------------------------------------------------
// Kernel 1: L2-normalize x_i / x_j rows, emit f16 z[8192,256], pos[], diag[].
// One wave (32 lanes) per pair; 8 floats per lane.
// ---------------------------------------------------------------------------
__global__ __launch_bounds__(128)
void contrast_normalize_kernel(const float* __restrict__ xi,
                               const float* __restrict__ xj,
                               _Float16* __restrict__ zh,
                               float* __restrict__ pos,
                               float* __restrict__ diag) {
    const int wave = threadIdx.x >> 5;
    const int lane = threadIdx.x & 31;
    const int k = blockIdx.x * 4 + wave;

    const float* pi = xi + (size_t)k * DIM + lane * 8;
    const float* pj = xj + (size_t)k * DIM + lane * 8;

    float a[8], b[8];
    {
        float4 a0 = ((const float4*)pi)[0], a1 = ((const float4*)pi)[1];
        float4 b0 = ((const float4*)pj)[0], b1 = ((const float4*)pj)[1];
        a[0]=a0.x; a[1]=a0.y; a[2]=a0.z; a[3]=a0.w;
        a[4]=a1.x; a[5]=a1.y; a[6]=a1.z; a[7]=a1.w;
        b[0]=b0.x; b[1]=b0.y; b[2]=b0.z; b[3]=b0.w;
        b[4]=b1.x; b[5]=b1.y; b[6]=b1.z; b[7]=b1.w;
    }

    float ssi = 0.f, ssj = 0.f;
    #pragma unroll
    for (int t = 0; t < 8; ++t) { ssi += a[t]*a[t]; ssj += b[t]*b[t]; }
    #pragma unroll
    for (int m = 16; m; m >>= 1) {      // full-wave (32 lane) reduction
        ssi += __shfl_xor(ssi, m, 32);
        ssj += __shfl_xor(ssj, m, 32);
    }

    const float invi = 1.0f / fmaxf(sqrtf(ssi), 1e-12f);
    const float invj = 1.0f / fmaxf(sqrtf(ssj), 1e-12f);

    _Float16* zi = zh + (size_t)k * DIM + lane * 8;
    _Float16* zj = zh + (size_t)(k + B_PAIRS) * DIM + lane * 8;

    float dot = 0.f;
    v8h hi16, hj16;
    #pragma unroll
    for (int t = 0; t < 8; ++t) {
        float zif = a[t] * invi;
        float zjf = b[t] * invj;
        dot += zif * zjf;
        hi16[t] = (_Float16)zif;
        hj16[t] = (_Float16)zjf;
    }
    *(v8h*)zi = hi16;
    *(v8h*)zj = hj16;

    #pragma unroll
    for (int m = 16; m; m >>= 1) dot += __shfl_xor(dot, m, 32);

    if (lane == 0) {
        pos[k]              = dot * INV_TEMP;
        diag[k]             = (ssi * invi * invi) * INV_TEMP;
        diag[k + B_PAIRS]   = (ssj * invj * invj) * INV_TEMP;
    }
}

// ---------------------------------------------------------------------------
// Kernel 2: fused Gram + exp + row-sum.
// Each wave owns rows [i0, i0+32): two 16-row A tiles register-resident
// (16 x v16h = 128 VGPRs), so every B tile fetched feeds TWO wmma chains
// (16 wmma per 16 b128 loads -> 32 flops/byte of L2 traffic).
// C/D layout: VGPR v, lanes 0-15 -> row v; lanes 16-31 -> row v+8.
// ---------------------------------------------------------------------------
__global__ __launch_bounds__(256, 1)
void contrast_gram_rowsum_kernel(const _Float16* __restrict__ zh,
                                 float* __restrict__ S) {
    const int lane  = threadIdx.x & 31;
    const int wave  = threadIdx.x >> 5;
    const int i0    = (blockIdx.x * 8 + wave) * 32;
    const int rlo   = lane & 15;
    const int khalf = (lane >> 4) & 1;

    // ---- A matrices: rows i0..i0+15 and i0+16..i0+31, K=0..255 ----
    // ISA 16-bit A layout: lanes 0-15 hold K={0..7,16..23}, lanes 16-31 K={8..15,24..31}
    v16h A0[8], A1[8];
    const _Float16* arow0 = zh + (size_t)(i0 + rlo) * DIM;
    const _Float16* arow1 = zh + (size_t)(i0 + 16 + rlo) * DIM;
    #pragma unroll
    for (int kk = 0; kk < 8; ++kk) {
        const _Float16* p0 = arow0 + kk * 32 + khalf * 8;
        const _Float16* p1 = arow1 + kk * 32 + khalf * 8;
        v8h lo0 = *(const v8h*)p0, hi0 = *(const v8h*)(p0 + 16);
        v8h lo1 = *(const v8h*)p1, hi1 = *(const v8h*)(p1 + 16);
        #pragma unroll
        for (int t = 0; t < 8; ++t) {
            A0[kk][t] = lo0[t]; A0[kk][t + 8] = hi0[t];
            A1[kk][t] = lo1[t]; A1[kk][t + 8] = hi1[t];
        }
    }

    float rowsum0[8], rowsum1[8];
    #pragma unroll
    for (int v = 0; v < 8; ++v) { rowsum0[v] = 0.f; rowsum1[v] = 0.f; }

    for (int j0 = 0; j0 < N2; j0 += 16) {
        const _Float16* brow = zh + (size_t)(j0 + rlo) * DIM;
        __builtin_prefetch(brow + 16 * DIM, 0, 3);   // next column tile, near-cache hint

        v8f c0 = {}, c1 = {};
        #pragma unroll
        for (int kk = 0; kk < 8; ++kk) {
            // B (32x16): column N = j0 + rlo, K={0..7,16..23} / {8..15,24..31} per half-wave
            const _Float16* p = brow + kk * 32 + khalf * 8;
            v8h lo = *(const v8h*)p;
            v8h hi = *(const v8h*)(p + 16);
            v16h bm;
            #pragma unroll
            for (int t = 0; t < 8; ++t) { bm[t] = lo[t]; bm[t + 8] = hi[t]; }
            c0 = __builtin_amdgcn_wmma_f32_16x16x32_f16(
                     false, A0[kk], false, bm, (short)0, c0, false, false);
            c1 = __builtin_amdgcn_wmma_f32_16x16x32_f16(
                     false, A1[kk], false, bm, (short)0, c1, false, false);
        }

        #pragma unroll
        for (int v = 0; v < 8; ++v) {
            rowsum0[v] += __expf(c0[v] * INV_TEMP);
            rowsum1[v] += __expf(c1[v] * INV_TEMP);
        }
    }

    // Reduce across the 16 lanes that share each row, then store.
    #pragma unroll
    for (int v = 0; v < 8; ++v) {
        float s0 = rowsum0[v];
        float s1 = rowsum1[v];
        #pragma unroll
        for (int m = 8; m; m >>= 1) {
            s0 += __shfl_xor(s0, m, 16);
            s1 += __shfl_xor(s1, m, 16);
        }
        if (rlo == 0) {
            S[i0 + v + 8 * khalf]      = s0;
            S[i0 + 16 + v + 8 * khalf] = s1;
        }
    }
}

// ---------------------------------------------------------------------------
// Kernel 3: loss_i = log(exp(p_i) + S_i - exp(diag_i)) - p_i ; write mean.
// ---------------------------------------------------------------------------
__global__ __launch_bounds__(1024)
void contrast_finalize_kernel(const float* __restrict__ pos,
                              const float* __restrict__ diag,
                              const float* __restrict__ S,
                              float* __restrict__ out) {
    __shared__ float sm[1024];
    const int t = threadIdx.x;
    float acc = 0.f;
    for (int i = t; i < N2; i += 1024) {
        float p = pos[i & (B_PAIRS - 1)];
        float denom = __expf(p) + S[i] - __expf(diag[i]);
        acc += __logf(denom) - p;
    }
    sm[t] = acc;
    __syncthreads();
    #pragma unroll
    for (int s = 512; s; s >>= 1) {
        if (t < s) sm[t] += sm[t + s];
        __syncthreads();
    }
    if (t == 0) out[0] = sm[0] * (1.0f / (float)N2);
}

// ---------------------------------------------------------------------------
extern "C" void kernel_launch(void* const* d_in, const int* in_sizes, int n_in,
                              void* d_out, int out_size, void* d_ws, size_t ws_size,
                              hipStream_t stream) {
    (void)in_sizes; (void)n_in; (void)out_size; (void)ws_size;

    const float* xi = (const float*)d_in[0];
    const float* xj = (const float*)d_in[1];
    float* out = (float*)d_out;

    // workspace layout
    _Float16* zh  = (_Float16*)d_ws;                                  // 8192*256*2 = 4 MB
    float*    pos = (float*)((char*)d_ws + (size_t)N2 * DIM * 2);     // 16 KB
    float*    dia = pos + B_PAIRS;                                    // 32 KB
    float*    S   = dia + N2;                                         // 32 KB

    contrast_normalize_kernel<<<B_PAIRS / 4, 128, 0, stream>>>(xi, xj, zh, pos, dia);
    contrast_gram_rowsum_kernel<<<N2 / (32 * 8), 256, 0, stream>>>(zh, S);
    contrast_finalize_kernel<<<1, 1024, 0, stream>>>(pos, dia, S, out);
}
